// ChunkedSurpriseGatedSSD_32959579030375
// MI455X (gfx1250) — matrix operand order repaired and checked by reading, not compiled, not run on hardware
//
#include <hip/hip_runtime.h>

// ---------------------------------------------------------------------------
// ChunkedSurpriseGatedSSD for MI455X (gfx1250, wave32, WMMA).
// Shapes fixed by reference: b=2, L=4096, H=16, dh=ds=64, CHUNK=64, NCHUNK=64.
// fp32 I/O; matmuls on v_wmma_f32_16x16x32_bf16 with hi/lo bf16 split
// (3 WMMAs per logical product) for near-fp32 accuracy.
// LDS is staged pre-packed in WMMA-operand order:
//   - B-operands: fragment-major, each lane's 8 words contiguous -> one 32B
//     vector load (2x ds_load_b128) fills the fragment VGPRs directly.
//   - A-operands: bf16-pair-per-row, stride 36 -> two aligned 16B runs.
// ---------------------------------------------------------------------------

typedef __attribute__((ext_vector_type(16))) __bf16        v16bf;
typedef __attribute__((ext_vector_type(2)))  __bf16        v2bf;
typedef __attribute__((ext_vector_type(8)))  float         v8f;
typedef __attribute__((ext_vector_type(8)))  unsigned int  v8u;
typedef __attribute__((ext_vector_type(4)))  unsigned int  v4u;

#define NB_   2
#define LQ_   4096
#define HH_   16
#define DH_   64
#define CH_   64
#define NCH_  64
#define ROWSTRIDE_ (HH_ * DH_)   // 1024 floats between consecutive L rows

#define CP_S  36   // A-operand pair-per-row stride (u32 words; 16B-aligned runs)
#define M_S   66   // fp32 M strip row stride (even: 8B-aligned pairs, bank-clean)

static __device__ inline v8f wmma_bf16(const v16bf& a, const v16bf& b, const v8f& c) {
  return __builtin_amdgcn_wmma_f32_16x16x32_bf16(false, a, false, b, (short)0, c,
                                                 false, false);
}

static __device__ inline v8f vzero8() {
  v8f z = {0.f, 0.f, 0.f, 0.f, 0.f, 0.f, 0.f, 0.f};
  return z;
}

static __device__ inline void splitf(float x, __bf16& hi, __bf16& lo) {
  hi = (__bf16)x;
  lo = (__bf16)(x - (float)hi);
}

// split two fp32 into packed bf16-pair words (hi parts, lo parts)
static __device__ inline void split_pack(float a, float b,
                                         unsigned int& ph, unsigned int& pl) {
  __bf16 h0, l0, h1, l1;
  splitf(a, h0, l0);
  splitf(b, h1, l1);
  v2bf vh = {h0, h1};
  v2bf vl = {l0, l1};
  ph = __builtin_bit_cast(unsigned int, vh);
  pl = __builtin_bit_cast(unsigned int, vl);
}

// ---- fragment-major B-operand layout ---------------------------------------
// Logical B-source element (k in [0,64), n in [0,64)), pair word = (k/2, n).
// word index = ((n>>4)*2 + (k>>5))*256 + lane*8 + r, lane = bit3(k)*16 + (n&15),
// r = ((k&7)>>1) + 4*bit4(k), halves = k parity.  Matches the WMMA 32x16 bf16 B layout.
static __device__ inline int bop_word(int k2, int n) {
  const int k    = k2 << 1;
  const int fb   = ((n >> 4) << 1) | (k >> 5);
  const int lane = (((k >> 3) & 1) << 4) | (n & 15);
  const int r    = ((k >> 1) & 3) | (((k >> 4) & 1) << 2);
  return (fb * 32 + lane) * 8 + r;
}

// one fragment = one contiguous 32B chunk per lane
static __device__ inline v16bf load_bfrag_f(const unsigned int* base, int tj,
                                            int kbi, int lane) {
  const unsigned int* p = base + ((tj * 2 + kbi) * 32 + lane) * 8;
  v8u f = *(const v8u*)p;
  return __builtin_bit_cast(v16bf, f);
}

// scalar element access into a fragment-major array (for A-uses of B^T)
static __device__ inline __bf16 bop_elem(const unsigned int* base, int s, int c) {
  v2bf q = __builtin_bit_cast(v2bf, base[bop_word(s >> 1, c)]);
  return q[s & 1];
}

// ---- A-operand (16x32 bf16) from pair-per-row array: word[row][k/2] --------
// lanes 0-15: M=lane, K in {0..7,16..23}; lanes 16-31: K in {8..15,24..31}.
// 8 words = two contiguous, 16B-aligned 4-word runs.
static __device__ inline v16bf load_afrag_p(const unsigned int* base, int row,
                                            int rstride, int kb, int lane) {
  const int khalf2 = (lane >> 4) << 2;  // (khalf)/2: 0 or 4
  const unsigned int* p = base + row * rstride + (kb >> 1) + khalf2;
  v4u lo = *(const v4u*)p;
  v4u hi = *(const v4u*)(p + 8);
  v8u f = __builtin_shufflevector(lo, hi, 0, 1, 2, 3, 4, 5, 6, 7);
  return __builtin_bit_cast(v16bf, f);
}

// ---------------------------------------------------------------------------
// Kernel 1: per-chunk. grid = b*H*NCH blocks, 128 threads (4 waves).
// Wave w owns output rows [16w, 16w+16).
// ---------------------------------------------------------------------------
__global__ __launch_bounds__(128) void ssd_chunk_kernel(
    const float* __restrict__ X, const float* __restrict__ A,
    const float* __restrict__ B, const float* __restrict__ C,
    const float* __restrict__ l2ab, const float* __restrict__ l2b,
    const float* __restrict__ sema, float* __restrict__ Y,
    float* __restrict__ hfw, float* __restrict__ dcw, float* __restrict__ dfw) {
  __shared__ __align__(32) unsigned int xph[2048], xpl[2048];  // X  [k=c][n=d]
  __shared__ __align__(32) unsigned int bph[2048], bpl[2048];  // B^T [k=s][n=c]
  __shared__ __align__(32) float cm[4608];  // C pairs (u32) then fp32 M strips
  __shared__ float a_s[64], dfl[64], rdfl[64];
  __shared__ float ssq, s_df63;

  unsigned int* cph = (unsigned int*)cm;  // C pairs over s: word[i][s/2]
  unsigned int* cpl = cph + 64 * CP_S;

  const int t    = blockIdx.x;
  const int ni   = t & (NCH_ - 1);
  const int h    = (t >> 6) & (HH_ - 1);
  const int bi   = t >> 10;
  const int tid  = threadIdx.x;
  const int lane = tid & 31;
  const int wv   = tid >> 5;

  const size_t rowL = (size_t)bi * LQ_ + (size_t)ni * CH_;
  const size_t off0 = rowL * ROWSTRIDE_ + (size_t)h * DH_;

  // ---- stage inputs (fp32 -> packed bf16 hi/lo pairs)
  for (int idx = tid; idx < 2048; idx += 128) {
    unsigned int ph, pl;
    {  // X: pack rows (2a, 2a+1) at column d -> fragment-major
      const int a = idx >> 6, d = idx & 63;
      const float x0 = X[off0 + (size_t)(2 * a) * ROWSTRIDE_ + d];
      const float x1 = X[off0 + (size_t)(2 * a + 1) * ROWSTRIDE_ + d];
      split_pack(x0, x1, ph, pl);
      const int w = bop_word(a, d);
      xph[w] = ph;
      xpl[w] = pl;
    }
    {  // B^T as B-op source [k=s][n=c]: pack (B[c][2s2], B[c][2s2+1])
      const int s2 = idx & 31, c = idx >> 5;
      const float b0 = B[off0 + (size_t)c * ROWSTRIDE_ + 2 * s2];
      const float b1 = B[off0 + (size_t)c * ROWSTRIDE_ + 2 * s2 + 1];
      split_pack(b0, b1, ph, pl);
      const int w = bop_word(s2, c);
      bph[w] = ph;
      bpl[w] = pl;
    }
    {  // C: word[i][s/2] packs (C[i][2s2], C[i][2s2+1])
      const int s2 = idx & 31, i = idx >> 5;
      const float c0 = C[off0 + (size_t)i * ROWSTRIDE_ + 2 * s2];
      const float c1 = C[off0 + (size_t)i * ROWSTRIDE_ + 2 * s2 + 1];
      split_pack(c0, c1, ph, pl);
      cph[i * CP_S + s2] = ph;
      cpl[i * CP_S + s2] = pl;
    }
  }
  if (tid < 64) a_s[tid] = A[(rowL + tid) * HH_ + h];
  if (tid == 0) ssq = 0.0f;
  __syncthreads();

  const int mrow = wv * 16 + (lane & 15);
  const int ncl  = lane & 15;

  // ---- surprise: BX = B^T X (hi-only; tanh gate saturates, low prec OK)
  {
    v16bf a0, a1;
    {
      const int khalf = (lane >> 4) << 3;
#pragma unroll
      for (int r = 0; r < 8; ++r) {
        const int c0 = khalf + ((r < 4) ? (2 * r) : (8 + 2 * r));
        a0[2 * r]     = bop_elem(bph, mrow, c0);
        a0[2 * r + 1] = bop_elem(bph, mrow, c0 + 1);
        a1[2 * r]     = bop_elem(bph, mrow, 32 + c0);
        a1[2 * r + 1] = bop_elem(bph, mrow, 32 + c0 + 1);
      }
    }
    float lss = 0.0f;
#pragma unroll
    for (int tj = 0; tj < 4; ++tj) {
      v16bf b0 = load_bfrag_f(xph, tj, 0, lane);
      v16bf b1 = load_bfrag_f(xph, tj, 1, lane);
      v8f acc = vzero8();
      acc = wmma_bf16(a0, b0, acc);
      acc = wmma_bf16(a1, b1, acc);
#pragma unroll
      for (int r = 0; r < 8; ++r) lss += acc[r] * acc[r];
    }
    atomicAdd(&ssq, lss);
  }
  __syncthreads();

  // ---- gate + cumulative decay (scalar work; thread 0)
  if (tid == 0) {
    const float surprise = ssq * (1.0f / 4096.0f);
    const float norm = surprise / (sema[h] + 1e-6f);
    const float l2c  = fminf(fmaxf(l2ab[h], -3.32f), -0.015f);
    const float ab   = 1.0f - exp2f(l2c);
    const float beta = exp2f(fminf(fmaxf(l2b[h], -2.0f), 2.0f));
    const float boost = fmaxf(0.0f, tanhf(beta * norm));
    const float alpha = fminf(fmaxf(ab + (1.0f - ab) * boost, 0.01f), 0.999f);
    const float oma = 1.0f - alpha;
    float cs = 0.0f;
    for (int c = 0; c < 64; ++c) {
      cs += a_s[c] * oma;                 // A_cs (|cs| < 64, exp never overflows)
      const float e = expf(cs);
      dfl[c]  = e;                        // decay_from_start
      rdfl[c] = expf(-cs);
      dfw[(size_t)t * 64 + c] = e;        // stash for kernel 3
    }
    s_df63 = dfl[63];
    dcw[t] = dfl[63];                     // decay_chunk
  }

  // C A-fragments must be in registers before cm is overwritten with M strips
  v16bf cA0h = load_afrag_p(cph, mrow, CP_S, 0, lane);
  v16bf cA1h = load_afrag_p(cph, mrow, CP_S, 32, lane);
  v16bf cA0l = load_afrag_p(cpl, mrow, CP_S, 0, lane);
  v16bf cA1l = load_afrag_p(cpl, mrow, CP_S, 32, lane);
  __syncthreads();

  // ---- CB = C B^T, apply causal decay mask, write M strip (reuses cm)
#pragma unroll
  for (int tj = 0; tj < 4; ++tj) {
    v16bf b0h = load_bfrag_f(bph, tj, 0, lane);
    v16bf b1h = load_bfrag_f(bph, tj, 1, lane);
    v16bf b0l = load_bfrag_f(bpl, tj, 0, lane);
    v16bf b1l = load_bfrag_f(bpl, tj, 1, lane);
    v8f acc = vzero8();
    acc = wmma_bf16(cA0h, b0h, acc);
    acc = wmma_bf16(cA0h, b0l, acc);
    acc = wmma_bf16(cA0l, b0h, acc);
    acc = wmma_bf16(cA1h, b1h, acc);
    acc = wmma_bf16(cA1h, b1l, acc);
    acc = wmma_bf16(cA1l, b1h, acc);
    const int j = tj * 16 + ncl;
#pragma unroll
    for (int r = 0; r < 8; ++r) {
      const int il = r + ((lane >> 4) << 3);
      const int i  = wv * 16 + il;
      cm[wv * 16 * M_S + il * M_S + j] = (i >= j) ? acc[r] * dfl[i] * rdfl[j] : 0.0f;
    }
  }
  __syncthreads();

  // ---- Y_intra = M X
  v16bf m0h, m0l, m1h, m1l;
  {
    const int rloc  = lane & 15;
    const int khalf = (lane >> 4) << 3;
    const float* mr = &cm[wv * 16 * M_S + rloc * M_S];
#pragma unroll
    for (int r = 0; r < 8; ++r) {
      const int kk = khalf + ((r < 4) ? 2 * r : 8 + 2 * r);
      __bf16 hi, lo;
      const float v0 = mr[kk];
      const float v1 = mr[kk + 1];
      splitf(v0, hi, lo); m0h[2 * r] = hi;     m0l[2 * r] = lo;
      splitf(v1, hi, lo); m0h[2 * r + 1] = hi; m0l[2 * r + 1] = lo;
      const float w0 = mr[32 + kk];
      const float w1 = mr[32 + kk + 1];
      splitf(w0, hi, lo); m1h[2 * r] = hi;     m1l[2 * r] = lo;
      splitf(w1, hi, lo); m1h[2 * r + 1] = hi; m1l[2 * r + 1] = lo;
    }
  }
#pragma unroll
  for (int tj = 0; tj < 4; ++tj) {
    v16bf b0h = load_bfrag_f(xph, tj, 0, lane);
    v16bf b1h = load_bfrag_f(xph, tj, 1, lane);
    v16bf b0l = load_bfrag_f(xpl, tj, 0, lane);
    v16bf b1l = load_bfrag_f(xpl, tj, 1, lane);
    v8f acc = vzero8();
    acc = wmma_bf16(m0h, b0h, acc);
    acc = wmma_bf16(m0h, b0l, acc);
    acc = wmma_bf16(m0l, b0h, acc);
    acc = wmma_bf16(m1h, b1h, acc);
    acc = wmma_bf16(m1h, b1l, acc);
    acc = wmma_bf16(m1l, b1h, acc);
#pragma unroll
    for (int r = 0; r < 8; ++r) {
      const int i = wv * 16 + r + ((lane >> 4) << 3);
      const int d = tj * 16 + ncl;
      Y[off0 + (size_t)i * ROWSTRIDE_ + d] = acc[r];
    }
  }

  // ---- h_final = (w ⊙ B)^T X,  w[c] = exp(acs[63] - acs[c])
  v16bf p0h, p0l, p1h, p1l;
  {
    const int khalf = (lane >> 4) << 3;
    const float d63 = s_df63;
#pragma unroll
    for (int r = 0; r < 8; ++r) {
      const int kk = khalf + ((r < 4) ? 2 * r : 8 + 2 * r);
      __bf16 hi, lo;
      {
        const float b0 = (float)bop_elem(bph, mrow, kk) +
                         (float)bop_elem(bpl, mrow, kk);
        const float b1 = (float)bop_elem(bph, mrow, kk + 1) +
                         (float)bop_elem(bpl, mrow, kk + 1);
        const float p0 = d63 * rdfl[kk] * b0;
        const float p1 = d63 * rdfl[kk + 1] * b1;
        splitf(p0, hi, lo); p0h[2 * r] = hi;     p0l[2 * r] = lo;
        splitf(p1, hi, lo); p0h[2 * r + 1] = hi; p0l[2 * r + 1] = lo;
      }
      {
        const int k2 = 32 + kk;
        const float b0 = (float)bop_elem(bph, mrow, k2) +
                         (float)bop_elem(bpl, mrow, k2);
        const float b1 = (float)bop_elem(bph, mrow, k2 + 1) +
                         (float)bop_elem(bpl, mrow, k2 + 1);
        const float p0 = d63 * rdfl[k2] * b0;
        const float p1 = d63 * rdfl[k2 + 1] * b1;
        splitf(p0, hi, lo); p1h[2 * r] = hi;     p1l[2 * r] = lo;
        splitf(p1, hi, lo); p1h[2 * r + 1] = hi; p1l[2 * r + 1] = lo;
      }
    }
  }
#pragma unroll
  for (int tj = 0; tj < 4; ++tj) {
    v16bf b0h = load_bfrag_f(xph, tj, 0, lane);
    v16bf b1h = load_bfrag_f(xph, tj, 1, lane);
    v16bf b0l = load_bfrag_f(xpl, tj, 0, lane);
    v16bf b1l = load_bfrag_f(xpl, tj, 1, lane);
    v8f acc = vzero8();
    acc = wmma_bf16(p0h, b0h, acc);
    acc = wmma_bf16(p0h, b0l, acc);
    acc = wmma_bf16(p0l, b0h, acc);
    acc = wmma_bf16(p1h, b1h, acc);
    acc = wmma_bf16(p1h, b1l, acc);
    acc = wmma_bf16(p1l, b1h, acc);
#pragma unroll
    for (int r = 0; r < 8; ++r) {
      const int s = wv * 16 + r + ((lane >> 4) << 3);
      const int d = tj * 16 + ncl;
      hfw[(size_t)t * 4096 + s * 64 + d] = acc[r];
    }
  }
}

// ---------------------------------------------------------------------------
// Kernel 2: inter-chunk scan, elementwise over the 64x64 state.
// grid = b*H*4 blocks, 256 threads, 4 state elements per thread.
// In-place: hfw[n] (h_final) is replaced by h_inter[n] (carry entering n).
// ---------------------------------------------------------------------------
__global__ __launch_bounds__(256) void ssd_scan_kernel(
    float* __restrict__ hfw, const float* __restrict__ dcw) {
  const int bh  = blockIdx.x >> 2;
  const int sl  = blockIdx.x & 3;
  const int tid = threadIdx.x;
  const size_t base = (size_t)bh * NCH_ * 4096;
  const int e0 = sl * 1024 + tid;
  float carry[4] = {0.f, 0.f, 0.f, 0.f};
  for (int n = 0; n < NCH_; ++n) {
    const size_t off = base + (size_t)n * 4096;
    const float d = dcw[bh * NCH_ + n];
#pragma unroll
    for (int i = 0; i < 4; ++i) {
      const size_t idx = off + e0 + i * 256;
      const float v = hfw[idx];
      hfw[idx] = carry[i];
      carry[i] = d * carry[i] + v;
    }
  }
}

// ---------------------------------------------------------------------------
// Kernel 3: Y += diag(exp(A_cs)) * (C @ h_inter). grid = b*H*NCH, 128 threads.
// ---------------------------------------------------------------------------
__global__ __launch_bounds__(128) void ssd_inter_kernel(
    const float* __restrict__ C, const float* __restrict__ hws,
    const float* __restrict__ dfw, float* __restrict__ Y) {
  __shared__ __align__(32) unsigned int cph[64 * CP_S], cpl[64 * CP_S];
  __shared__ __align__(32) unsigned int hph[2048], hpl[2048];  // h [k=s][n=d]
  __shared__ float dfl[64];

  const int t    = blockIdx.x;
  const int ni   = t & (NCH_ - 1);
  const int h    = (t >> 6) & (HH_ - 1);
  const int bi   = t >> 10;
  const int tid  = threadIdx.x;
  const int lane = tid & 31;
  const int wv   = tid >> 5;

  const size_t rowL = (size_t)bi * LQ_ + (size_t)ni * CH_;
  const size_t off0 = rowL * ROWSTRIDE_ + (size_t)h * DH_;

  for (int idx = tid; idx < 2048; idx += 128) {
    unsigned int ph, pl;
    {  // C: word[i][s/2]
      const int s2 = idx & 31, i = idx >> 5;
      const float c0 = C[off0 + (size_t)i * ROWSTRIDE_ + 2 * s2];
      const float c1 = C[off0 + (size_t)i * ROWSTRIDE_ + 2 * s2 + 1];
      split_pack(c0, c1, ph, pl);
      cph[i * CP_S + s2] = ph;
      cpl[i * CP_S + s2] = pl;
    }
    {  // h: pack rows (2a, 2a+1) at column d -> fragment-major
      const int a = idx >> 6, d = idx & 63;
      const float h0 = hws[(size_t)t * 4096 + (2 * a) * 64 + d];
      const float h1 = hws[(size_t)t * 4096 + (2 * a + 1) * 64 + d];
      split_pack(h0, h1, ph, pl);
      const int w = bop_word(a, d);
      hph[w] = ph;
      hpl[w] = pl;
    }
  }
  if (tid < 64) dfl[tid] = dfw[(size_t)t * 64 + tid];
  __syncthreads();

  const int mrow = wv * 16 + (lane & 15);
  const int ncl  = lane & 15;

  v16bf a0h = load_afrag_p(cph, mrow, CP_S, 0, lane);
  v16bf a1h = load_afrag_p(cph, mrow, CP_S, 32, lane);
  v16bf a0l = load_afrag_p(cpl, mrow, CP_S, 0, lane);
  v16bf a1l = load_afrag_p(cpl, mrow, CP_S, 32, lane);

#pragma unroll
  for (int tj = 0; tj < 4; ++tj) {
    v16bf b0h = load_bfrag_f(hph, tj, 0, lane);
    v16bf b1h = load_bfrag_f(hph, tj, 1, lane);
    v16bf b0l = load_bfrag_f(hpl, tj, 0, lane);
    v16bf b1l = load_bfrag_f(hpl, tj, 1, lane);
    v8f acc = vzero8();
    acc = wmma_bf16(a0h, b0h, acc);
    acc = wmma_bf16(a0h, b0l, acc);
    acc = wmma_bf16(a0l, b0h, acc);
    acc = wmma_bf16(a1h, b1h, acc);
    acc = wmma_bf16(a1h, b1l, acc);
    acc = wmma_bf16(a1l, b1h, acc);
#pragma unroll
    for (int r = 0; r < 8; ++r) {
      const int i = wv * 16 + r + ((lane >> 4) << 3);
      const int d = tj * 16 + ncl;
      const size_t yi = off0 + (size_t)i * ROWSTRIDE_ + d;
      Y[yi] = Y[yi] + dfl[i] * acc[r];
    }
  }
}

// ---------------------------------------------------------------------------
extern "C" void kernel_launch(void* const* d_in, const int* in_sizes, int n_in,
                              void* d_out, int out_size, void* d_ws,
                              size_t ws_size, hipStream_t stream) {
  (void)in_sizes; (void)n_in; (void)out_size; (void)ws_size;
  const float* X    = (const float*)d_in[0];
  const float* A    = (const float*)d_in[1];
  const float* B    = (const float*)d_in[2];
  const float* C    = (const float*)d_in[3];
  const float* l2ab = (const float*)d_in[4];
  const float* l2b  = (const float*)d_in[5];
  const float* sema = (const float*)d_in[6];
  float* Y = (float*)d_out;

  const int NT = NB_ * HH_ * NCH_;          // 2048 chunk tiles
  float* hfw = (float*)d_ws;                // [NT][64][64]  h_final -> h_inter
  float* dcw = hfw + (size_t)NT * 4096;     // [NT]          decay_chunk
  float* dfw = dcw + NT;                    // [NT][64]      decay_from_start

  ssd_chunk_kernel<<<NT, 128, 0, stream>>>(X, A, B, C, l2ab, l2b, sema, Y, hfw,
                                           dcw, dfw);
  ssd_scan_kernel<<<NB_ * HH_ * 4, 256, 0, stream>>>(hfw, dcw);
  ssd_inter_kernel<<<NT, 128, 0, stream>>>(C, hfw, dfw, Y);
}
